// SingleViewRNAGNN_22093311771384
// MI455X (gfx1250) — matrix operand order repaired and checked by reading, not compiled, not run on hardware
//
#include <hip/hip_runtime.h>
#include <hip/hip_bf16.h>

#define DEVFN __device__ __forceinline__
#define CDIV(a, b) (((a) + (b) - 1) / (b))

typedef __attribute__((ext_vector_type(16))) __bf16 v16bf;
typedef __attribute__((ext_vector_type(8)))  float  v8f;

namespace {

constexpr int N_NODES  = 50000;
constexpr int N_EDGES  = 400000;
constexpr int EP       = N_EDGES + N_NODES;   // edges incl. self loops
constexpr int F_IN     = 128;
constexpr int HID      = 256;
constexpr int HEADS    = 4;
constexpr int OUTD     = 128;
constexpr int NGRAPH   = 64;
constexpr float NEG_SLOPE = 0.2f;

// ---------- helpers ----------

DEVFN void edge_ep(const int* __restrict__ ei, int e, int& s, int& d) {
    if (e < N_EDGES) { s = ei[e]; d = ei[N_EDGES + e]; }
    else             { s = e - N_EDGES; d = s; }       // self loop
}

DEVFN float lrelu(float x) { return x > 0.f ? x : NEG_SLOPE * x; }
DEVFN float elu(float x)   { return x > 0.f ? x : __expf(x) - 1.f; }

// order-preserving float<->uint encoding so we can use integer atomicMax
DEVFN unsigned fenc(float f) {
    unsigned u = __float_as_uint(f);
    return (u & 0x80000000u) ? ~u : (u | 0x80000000u);
}
DEVFN float fdec(unsigned u) {
    return __uint_as_float((u & 0x80000000u) ? (u & 0x7fffffffu) : ~u);
}

// ---------- zero fill ----------

__global__ void zero_f32(float* __restrict__ p, size_t n) {
    size_t t = (size_t)blockIdx.x * blockDim.x + threadIdx.x;
    if (t < n) p[t] = 0.f;
}

// ---------- weight pre-swizzle: f32 row-major [K,N] -> bf16 WMMA B fragments ----
// Layout: frag index f = (kstep*NT + ntile); element ((f*32 + lane)*16 + j).
// Per ISA B 32x16 bf16 layout: lane = 16*hi + col, VGPR p (=j/2) holds
// K = kstep*32 + 8*hi + 2p (+1) for p<4, +8 more for p>=4, for that column.
template<int K_TOT, int N_TOT>
__global__ void swizzle_w(const float* __restrict__ W, __bf16* __restrict__ out) {
    constexpr int KSTEPS = K_TOT / 32;
    constexpr int NT = N_TOT / 16;
    int t = blockIdx.x * blockDim.x + threadIdx.x;
    if (t >= KSTEPS * NT * 32) return;
    const int lane = t & 31;
    const int nt = (t >> 5) % NT;
    const int ks = (t >> 5) / NT;
    const int hi = lane >> 4, lm = lane & 15;
    __bf16* o = out + (size_t)t * 16;
    #pragma unroll
    for (int p = 0; p < 8; ++p) {
        const int kb = ks * 32 + 8 * hi + 2 * p + (p >= 4 ? 8 : 0);
        o[2 * p]     = (__bf16)W[(size_t)kb * N_TOT + nt * 16 + lm];
        o[2 * p + 1] = (__bf16)W[(size_t)(kb + 1) * N_TOT + nt * 16 + lm];
    }
}

// ---------- bf16 WMMA GEMM:  O[M,N] = X[M,K] @ W[K,N] ----------
// One wave -> one 16x64 strip: A fragment loaded once per K-step, then 4
// back-to-back v_wmma_f32_16x16x32_bf16 sharing it (4 f32 accumulators).
// B comes pre-swizzled in bf16 fragment layout -> 2x global_load_b128/lane.
// Block = N_TOT/64 waves, grid.x = M/16.
template<int K_TOT, int N_TOT>
__global__ void __launch_bounds__(128)
gemm_bf16_wmma(const float* __restrict__ X, const __bf16* __restrict__ Wsw,
               float* __restrict__ O) {
    constexpr int NT = N_TOT / 16;
    const int lane = threadIdx.x & 31;
    const int wave = threadIdx.x >> 5;
    const int hi   = lane >> 4;     // half-wave select
    const int lm   = lane & 15;     // row (A) / col (C/D)
    const int m0   = blockIdx.x * 16;
    const int nt0  = wave * 4;      // first of 4 N-tiles for this wave

    v8f acc[4] = {};
    const float* __restrict__ arow = X + (size_t)(m0 + lm) * K_TOT;
    const v16bf* __restrict__ wv = (const v16bf*)Wsw;

    for (int k0 = 0; k0 < K_TOT; k0 += 32) {
        const int ks = k0 >> 5;
        // A 16x32 bf16: VGPR p in 0..3 -> K = k0 + 8*hi + 2p (+1)
        //               VGPR p in 4..7 -> K = +16
        v16bf a;
        #pragma unroll
        for (int p = 0; p < 4; ++p) {
            const int ka = k0 + 8 * hi + 2 * p;
            a[2 * p]         = (__bf16)arow[ka];
            a[2 * p + 1]     = (__bf16)arow[ka + 1];
            a[8 + 2 * p]     = (__bf16)arow[ka + 16];
            a[8 + 2 * p + 1] = (__bf16)arow[ka + 17];
        }
        #pragma unroll
        for (int u = 0; u < 4; ++u) {
            const v16bf b = wv[((size_t)ks * NT + (nt0 + u)) * 32 + lane];
            acc[u] = __builtin_amdgcn_wmma_f32_16x16x32_bf16(
                /*neg_a=*/false, a, /*neg_b=*/false, b,
                /*c_mod=*/(short)0, acc[u], /*reuse_a=*/false, /*reuse_b=*/false);
        }
    }

    // C/D f32 16x16: lane holds rows m0 + 8*hi + j, col n0 + lm
    #pragma unroll
    for (int u = 0; u < 4; ++u) {
        const int n0 = (nt0 + u) * 16;
        #pragma unroll
        for (int j = 0; j < 8; ++j)
            O[(size_t)(m0 + 8 * hi + j) * N_TOT + n0 + lm] = acc[u][j];
    }
}

// ---------- attention coefficients: a_s[i,h] = <xw[i,h,:], att_src[h,:]> ----------

template<int H, int CHAN>
__global__ void attn_coef(const float* __restrict__ xw,
                          const float* __restrict__ atts,
                          const float* __restrict__ attd,
                          float* __restrict__ os, float* __restrict__ od) {
    constexpr int C = CHAN / H;
    int t = blockIdx.x * blockDim.x + threadIdx.x;
    if (t >= N_NODES * H) return;
    const int i = t / H, h = t % H;
    const float* __restrict__ row = xw + (size_t)i * CHAN + h * C;
    const float* __restrict__ s = atts + h * C;
    const float* __restrict__ d = attd + h * C;
    float accs = 0.f, accd = 0.f;
    #pragma unroll 8
    for (int c = 0; c < C; ++c) {
        const float v = row[c];
        accs += v * s[c];
        accd += v * d[c];
    }
    os[t] = accs;
    od[t] = accd;
}

// ---------- edge pass 1: segment-max of leaky_relu scores ----------

template<int H>
__global__ void edge_max(const int* __restrict__ ei,
                         const float* __restrict__ as_, const float* __restrict__ ad_,
                         unsigned* __restrict__ m) {
    int e = blockIdx.x * blockDim.x + threadIdx.x;
    if (e >= EP) return;
    int s, d; edge_ep(ei, e, s, d);
    #pragma unroll
    for (int h = 0; h < H; ++h) {
        const float v = lrelu(as_[s * H + h] + ad_[d * H + h]);
        atomicMax(m + (size_t)d * H + h, fenc(v));
    }
}

// ---------- edge pass 2: ee = exp(score - max), segment-sum denominators ----------

template<int H>
__global__ void edge_expsum(const int* __restrict__ ei,
                            const float* __restrict__ as_, const float* __restrict__ ad_,
                            const unsigned* __restrict__ m,
                            float* __restrict__ ee, float* __restrict__ dn) {
    int e = blockIdx.x * blockDim.x + threadIdx.x;
    if (e >= EP) return;
    int s, d; edge_ep(ei, e, s, d);
    #pragma unroll
    for (int h = 0; h < H; ++h) {
        const float v  = lrelu(as_[s * H + h] + ad_[d * H + h]);
        const float mm = fdec(m[(size_t)d * H + h]);
        const float w  = __expf(v - mm);
        ee[(size_t)e * H + h] = w;
        atomicAdd(dn + (size_t)d * H + h, w);
    }
}

// ---------- edge pass 3: agg[dst] += alpha * xw[src]  (float4 vectorized) ----------

template<int H, int CHAN>
__global__ void edge_agg(const int* __restrict__ ei,
                         const float* __restrict__ xw,
                         const float* __restrict__ ee, const float* __restrict__ dn,
                         float* __restrict__ agg) {
    constexpr int V4 = CHAN / 4;
    int t = blockIdx.x * blockDim.x + threadIdx.x;
    if (t >= EP * V4) return;
    const int e = t / V4, q = t % V4;
    int s, d; edge_ep(ei, e, s, d);
    const int h = (q * 4) / (CHAN / H);
    const float alpha = ee[(size_t)e * H + h] / dn[(size_t)d * H + h];
    const float4 v = ((const float4*)xw)[(size_t)s * V4 + q];
    float* o = agg + (size_t)d * CHAN + q * 4;
    atomicAdd(o + 0, v.x * alpha);
    atomicAdd(o + 1, v.y * alpha);
    atomicAdd(o + 2, v.z * alpha);
    atomicAdd(o + 3, v.w * alpha);
}

// ---------- bias + ELU (in place) ----------

__global__ void bias_elu(float* __restrict__ buf, const float* __restrict__ bias,
                         int ncols, size_t total) {
    size_t t = (size_t)blockIdx.x * blockDim.x + threadIdx.x;
    if (t >= total) return;
    const int c = (int)(t % ncols);
    buf[t] = elu(buf[t] + bias[c]);
}

// ---------- final: h = ELU(agg + b2), pool sums + counts ----------

__global__ void final_elu_pool(float* __restrict__ hout, const float* __restrict__ b2,
                               const int* __restrict__ batch,
                               float* __restrict__ pool, float* __restrict__ counts) {
    int t = blockIdx.x * blockDim.x + threadIdx.x;
    if (t >= N_NODES * OUTD) return;
    const int i = t / OUTD, c = t % OUTD;
    const float x = elu(hout[t] + b2[c]);
    hout[t] = x;
    const int g = batch[i];
    atomicAdd(pool + (size_t)g * OUTD + c, x);
    if (c == 0) atomicAdd(counts + g, 1.0f);
}

__global__ void pool_div(const float* __restrict__ pool, const float* __restrict__ counts,
                         float* __restrict__ gemb) {
    int t = blockIdx.x * blockDim.x + threadIdx.x;
    if (t >= NGRAPH * OUTD) return;
    gemb[t] = pool[t] / fmaxf(counts[t / OUTD], 1.0f);
}

} // anonymous namespace

extern "C" void kernel_launch(void* const* d_in, const int* in_sizes, int n_in,
                              void* d_out, int out_size, void* d_ws, size_t ws_size,
                              hipStream_t stream) {
    (void)in_sizes; (void)n_in; (void)out_size; (void)ws_size;

    const float* x        = (const float*)d_in[0];   // [N, 128]
    const int*   ei       = (const int*)  d_in[1];   // [2, E]
    const int*   batch    = (const int*)  d_in[2];   // [N]
    const float* W1       = (const float*)d_in[3];   // [128, 256]
    const float* att_src1 = (const float*)d_in[4];   // [4, 64]
    const float* att_dst1 = (const float*)d_in[5];   // [4, 64]
    const float* b1       = (const float*)d_in[6];   // [256]
    const float* W2       = (const float*)d_in[7];   // [256, 128]
    const float* att_src2 = (const float*)d_in[8];   // [1, 128]
    const float* att_dst2 = (const float*)d_in[9];   // [1, 128]
    const float* b2       = (const float*)d_in[10];  // [128]

    float* gemb = (float*)d_out;                     // [G, 128]
    float* hout = gemb + (size_t)NGRAPH * OUTD;      // [N, 128]

    // ---- workspace layout (floats) ----
    float* ws  = (float*)d_ws;
    float* xw1 = ws;                                   // N*HID
    float* xw2 = xw1 + (size_t)N_NODES * HID;          // N*OUTD
    float* as1 = xw2 + (size_t)N_NODES * OUTD;         // N*HEADS
    float* ad1 = as1 + (size_t)N_NODES * HEADS;        // N*HEADS
    float* ee1 = ad1 + (size_t)N_NODES * HEADS;        // EP*HEADS
    float* as2 = ee1 + (size_t)EP * HEADS;             // N
    float* ad2 = as2 + N_NODES;                        // N
    float* ee2 = ad2 + N_NODES;                        // EP
    // zero-initialized group (contiguous):
    float*    zgrp   = ee2 + EP;
    float*    agg1   = zgrp;                           // N*HID
    float*    dn1    = agg1 + (size_t)N_NODES * HID;   // N*HEADS
    unsigned* m1     = (unsigned*)(dn1 + (size_t)N_NODES * HEADS); // N*HEADS
    float*    dn2    = (float*)(m1 + (size_t)N_NODES * HEADS);     // N
    unsigned* m2     = (unsigned*)(dn2 + N_NODES);     // N
    float*    pool   = (float*)(m2 + N_NODES);         // G*OUTD
    float*    counts = pool + (size_t)NGRAPH * OUTD;   // G
    const size_t zcount = (size_t)((counts + NGRAPH) - zgrp);
    // bf16 swizzled weights (32B-aligned for v16bf loads)
    size_t wboff = (size_t)((counts + NGRAPH) - ws);
    wboff = (wboff + 15u) & ~(size_t)15u;
    __bf16* wb1 = (__bf16*)(ws + wboff);               // (128/32)*(256/16)*32*16 bf16
    __bf16* wb2 = wb1 + (size_t)(F_IN / 32) * (HID / 16) * 32 * 16;
    constexpr int WB1_FRAGS = (F_IN / 32) * (HID / 16) * 32;   // lane-frags
    constexpr int WB2_FRAGS = (HID / 32) * (OUTD / 16) * 32;

    const int B = 256;

    // ---- init accumulators (every call: harness poisons ws/out) ----
    zero_f32<<<CDIV(zcount, (size_t)B), B, 0, stream>>>(zgrp, zcount);
    zero_f32<<<CDIV((size_t)N_NODES * OUTD, (size_t)B), B, 0, stream>>>(
        hout, (size_t)N_NODES * OUTD);

    // ---- pre-swizzle weights into bf16 WMMA fragment layout ----
    swizzle_w<F_IN, HID><<<CDIV(WB1_FRAGS, B), B, 0, stream>>>(W1, wb1);
    swizzle_w<HID, OUTD><<<CDIV(WB2_FRAGS, B), B, 0, stream>>>(W2, wb2);

    // ---- layer 0: xw1 = x @ W1 (bf16 WMMA, f32 accum) ----
    gemm_bf16_wmma<F_IN, HID><<<N_NODES / 16, 32 * (HID / 64), 0, stream>>>(x, wb1, xw1);
    attn_coef<HEADS, HID><<<CDIV(N_NODES * HEADS, B), B, 0, stream>>>(
        xw1, att_src1, att_dst1, as1, ad1);
    edge_max<HEADS><<<CDIV(EP, B), B, 0, stream>>>(ei, as1, ad1, m1);
    edge_expsum<HEADS><<<CDIV(EP, B), B, 0, stream>>>(ei, as1, ad1, m1, ee1, dn1);
    edge_agg<HEADS, HID><<<CDIV(EP * (HID / 4), B), B, 0, stream>>>(ei, xw1, ee1, dn1, agg1);
    bias_elu<<<CDIV((size_t)N_NODES * HID, (size_t)B), B, 0, stream>>>(
        agg1, b1, HID, (size_t)N_NODES * HID);

    // ---- layer 1: xw2 = h1 @ W2 (bf16 WMMA, f32 accum) ----
    gemm_bf16_wmma<HID, OUTD><<<N_NODES / 16, 32 * (OUTD / 64), 0, stream>>>(agg1, wb2, xw2);
    attn_coef<1, OUTD><<<CDIV(N_NODES, B), B, 0, stream>>>(
        xw2, att_src2, att_dst2, as2, ad2);
    edge_max<1><<<CDIV(EP, B), B, 0, stream>>>(ei, as2, ad2, m2);
    edge_expsum<1><<<CDIV(EP, B), B, 0, stream>>>(ei, as2, ad2, m2, ee2, dn2);
    edge_agg<1, OUTD><<<CDIV(EP * (OUTD / 4), B), B, 0, stream>>>(ei, xw2, ee2, dn2, hout);

    // ---- ELU + graph mean pooling ----
    final_elu_pool<<<CDIV(N_NODES * OUTD, B), B, 0, stream>>>(hout, b2, batch, pool, counts);
    pool_div<<<CDIV(NGRAPH * OUTD, B), B, 0, stream>>>(pool, counts, gemb);
}
